// RadialCTC_91182155694749
// MI455X (gfx1250) — matrix operand order repaired and checked by reading, not compiled, not run on hardware
//
#include <hip/hip_runtime.h>

// RadialCTC on MI455X (gfx1250):
//   bf16 WMMA GEMM with TDM (tensor_load_to_lds) double-buffered LDS staging,
//   f32 norm folding, log-softmax, CTC forward scan, deterministic sum.

#define TT 512
#define NNB 32
#define CC 1296
#define DD 512
#define SSL 30
#define LLS 61            // 2*S+1
#define NEG_INF (-1e9f)

#define KC 128            // K-chunk staged per TDM transfer
#define NKC (DD / KC)     // 4 chunks
#define BM 64             // block M tile (4 waves x 16)
#define BN 48             // block N tile (3 x 16 per wave)

typedef __attribute__((ext_vector_type(16))) __bf16 v16bf;
typedef __attribute__((ext_vector_type(8)))  __bf16 v8bf;
typedef __attribute__((ext_vector_type(8)))  float  v8f;
typedef __attribute__((ext_vector_type(4)))  unsigned int u32x4;
typedef __attribute__((ext_vector_type(8)))  int i32x8;
typedef __attribute__((ext_vector_type(4)))  int i32x4;

#if defined(__has_builtin)
#if __has_builtin(__builtin_amdgcn_tensor_load_to_lds) && __has_builtin(__builtin_amdgcn_s_wait_tensorcnt)
#define USE_TDM 1
#endif
#endif
#ifndef USE_TDM
#define USE_TDM 0
#endif

// ---------------- K1: per-row feature norms + f32 -> bf16 ----------------
__global__ void k_feat_norm(const float* __restrict__ feats,
                            __bf16* __restrict__ fB,
                            float* __restrict__ rinv) {
  const int r    = blockIdx.x;          // 0 .. T*N-1
  const int lane = threadIdx.x;         // wave32
  const float* src = feats + (size_t)r * DD;
  __bf16*      dst = fB    + (size_t)r * DD;
  float ss = 0.f;
#pragma unroll
  for (int i = 0; i < DD / 32; ++i) {
    const int d = lane + 32 * i;
    const float v = src[d];
    ss += v * v;
    dst[d] = (__bf16)v;
  }
#pragma unroll
  for (int m = 16; m > 0; m >>= 1) ss += __shfl_xor(ss, m, 32);
  if (lane == 0) rinv[r] = rsqrtf(ss);
}

// --------- K2: per-column weight norms + transpose W(D,C) -> Wt(C,D) bf16 ---------
__global__ void k_w_norm(const float* __restrict__ w,
                         __bf16* __restrict__ wT,
                         float* __restrict__ cinv) {
  const int c    = blockIdx.x;          // 0 .. C-1
  const int lane = threadIdx.x;         // wave32
  float ss = 0.f;
#pragma unroll
  for (int i = 0; i < DD / 32; ++i) {
    const int d = lane + 32 * i;
    const float v = w[(size_t)d * CC + c];
    ss += v * v;
    wT[(size_t)c * DD + d] = (__bf16)v;
  }
#pragma unroll
  for (int m = 16; m > 0; m >>= 1) ss += __shfl_xor(ss, m, 32);
  if (lane == 0) cinv[c] = rsqrtf(ss);
}

#if USE_TDM
// Build a 2D D# and issue TENSOR_LOAD_TO_LDS (TDM). All args uniform.
// Group0: [1:0]=count=1 | [63:32]=lds_addr | [120:64]=global_addr | [127:126]=type=2
// Group1: [17:16]=data_size(1 -> 2B) | tensor_dim0/1 | tile_dim0/1 | tensor_dim0_stride
__device__ __forceinline__ void tdm_load_tile_2d(unsigned lds_off,
                                                 unsigned long long gaddr,
                                                 unsigned td0, unsigned td1,
                                                 unsigned tl0, unsigned tl1,
                                                 unsigned long long st0) {
  union { u32x4 v; unsigned a[4]; } g0;
  g0.a[0] = 1u;                                             // count=1 (user descriptor)
  g0.a[1] = lds_off;                                        // lds_addr (bytes)
  g0.a[2] = (unsigned)gaddr;                                // global_addr[31:0]
  g0.a[3] = ((unsigned)(gaddr >> 32) & 0x01FFFFFFu) | (2u << 30);  // [56:32] | type=2
  union { i32x8 v; unsigned a[8]; } g1;
  g1.a[0] = 1u << 16;                                       // data_size=1 -> 2 bytes
  g1.a[1] = (td0 & 0xFFFFu) << 16;                          // tensor_dim0[15:0]
  g1.a[2] = (td0 >> 16) | ((td1 & 0xFFFFu) << 16);          // tensor_dim0[31:16]|dim1[15:0]
  g1.a[3] = (td1 >> 16) | (tl0 << 16);                      // tensor_dim1[31:16]|tile_dim0
  g1.a[4] = tl1 & 0xFFFFu;                                  // tile_dim1 (tile_dim2=0)
  g1.a[5] = (unsigned)st0;                                  // tensor_dim0_stride[31:0]
  g1.a[6] = (unsigned)(st0 >> 32) & 0xFFFFu;                // stride[47:32] (dim1_stride=0)
  g1.a[7] = 0u;
  i32x4 z4 = {0, 0, 0, 0};
#if __clang_major__ >= 23
  i32x8 z8 = {0, 0, 0, 0, 0, 0, 0, 0};
  __builtin_amdgcn_tensor_load_to_lds(g0.v, g1.v, z4, z4, z8, 0);
#else
  __builtin_amdgcn_tensor_load_to_lds(g0.v, g1.v, z4, z4, 0);
#endif
}
#endif

// ---------------- K3: raw GEMM out = feats_bf16 @ Wt^T via WMMA + TDM staging ----------------
// 128-thread block (4 waves) computes a 64x48 tile; K staged in 128-wide chunks,
// double-buffered in LDS via the Tensor Data Mover; each wave holds 3 accumulators.
__global__ void __launch_bounds__(128)
k_gemm_wmma(const __bf16* __restrict__ A,   // (T*N, D) row-major
            const __bf16* __restrict__ Bt,  // (C, D)   col-of-W contiguous
            float* __restrict__ out) {      // (T*N, C) raw dot products
  __shared__ __align__(16) __bf16 As[2][BM * KC];   // 2 x 16 KB
  __shared__ __align__(16) __bf16 Bs[2][BN * KC];   // 2 x 12 KB
  const int tid  = threadIdx.x;
  const int wave = tid >> 5;
  const int lane = tid & 31;
  const int half = lane >> 4;           // 0 | 1
  const int l16  = lane & 15;
  const int M0   = blockIdx.y * BM;     // gridDim.y = 256
  const int N0   = blockIdx.x * BN;     // gridDim.x = 27

#if USE_TDM
  const unsigned long long abase = (unsigned long long)(uintptr_t)A;
  const unsigned long long bbase = (unsigned long long)(uintptr_t)Bt;
  auto issue_chunk = [&](int c, int st) {
    if (tid < 32) {   // wave 0 issues both TDM descriptors (EXEC ignored by TDM)
      tdm_load_tile_2d((unsigned)(uintptr_t)&As[st][0],
                       abase + (((unsigned long long)M0 * DD) + (unsigned)(c * KC)) * 2ull,
                       DD, (unsigned)(TT * NNB), KC, BM, DD);
      tdm_load_tile_2d((unsigned)(uintptr_t)&Bs[st][0],
                       bbase + (((unsigned long long)N0 * DD) + (unsigned)(c * KC)) * 2ull,
                       DD, CC, KC, BN, DD);
    }
  };
#else
  auto issue_chunk = [&](int c, int st) {  // fallback: cooperative vector copy
    for (int i = tid; i < BM * (KC / 8); i += 128) {
      const int row = i / (KC / 8), col = i % (KC / 8);
      *(v8bf*)&As[st][row * KC + col * 8] =
          *(const v8bf*)(A + (size_t)(M0 + row) * DD + c * KC + col * 8);
    }
    for (int i = tid; i < BN * (KC / 8); i += 128) {
      const int row = i / (KC / 8), col = i % (KC / 8);
      *(v8bf*)&Bs[st][row * KC + col * 8] =
          *(const v8bf*)(Bt + (size_t)(N0 + row) * DD + c * KC + col * 8);
    }
  };
#endif

  v8f acc0 = {}, acc1 = {}, acc2 = {};
  issue_chunk(0, 0);

  for (int c = 0; c < NKC; ++c) {
#if USE_TDM
    if (c + 1 < NKC) {
      issue_chunk(c + 1, (c + 1) & 1);                  // buffer consumed 2 iters ago
      if (tid < 32) __builtin_amdgcn_s_wait_tensorcnt(2);  // oldest pair (chunk c) done
    } else {
      if (tid < 32) __builtin_amdgcn_s_wait_tensorcnt(0);
    }
#else
    if (c + 1 < NKC) issue_chunk(c + 1, (c + 1) & 1);
#endif
    __syncthreads();                                    // publish LDS chunk c

    const int st = c & 1;
    const __bf16* arow = &As[st][(wave * 16 + l16) * KC];   // A: lane holds row M=l16
    const __bf16* b0   = &Bs[st][(l16) * KC];               // B: lane holds column N=l16
    const __bf16* b1   = &Bs[st][(16 + l16) * KC];
    const __bf16* b2   = &Bs[st][(32 + l16) * KC];
#pragma unroll
    for (int j = 0; j < KC / 32; ++j) {
      const int k0 = j * 32;
      // A 16x32 (contiguous, matches B): half 0 -> K 0..15; half 1 -> K 16..31
      union { v16bf v; v8bf h[2]; } a, b;
      a.h[0] = *(const v8bf*)(arow + k0 + half * 16);
      a.h[1] = *(const v8bf*)(arow + k0 + half * 16 + 8);
      // B 32x16 (contiguous, matches A): lanes 0-15 K=k0..k0+15, lanes 16-31 K=k0+16..k0+31
      b.h[0] = *(const v8bf*)(b0 + k0 + half * 16);
      b.h[1] = *(const v8bf*)(b0 + k0 + half * 16 + 8);
      acc0 = __builtin_amdgcn_wmma_f32_16x16x32_bf16(false, a.v, false, b.v,
                                                     (short)0, acc0, false, false);
      b.h[0] = *(const v8bf*)(b1 + k0 + half * 16);
      b.h[1] = *(const v8bf*)(b1 + k0 + half * 16 + 8);
      acc1 = __builtin_amdgcn_wmma_f32_16x16x32_bf16(false, a.v, false, b.v,
                                                     (short)0, acc1, false, false);
      b.h[0] = *(const v8bf*)(b2 + k0 + half * 16);
      b.h[1] = *(const v8bf*)(b2 + k0 + half * 16 + 8);
      acc2 = __builtin_amdgcn_wmma_f32_16x16x32_bf16(false, a.v, false, b.v,
                                                     (short)0, acc2, false, false);
    }
    __syncthreads();                                    // chunk c consumed
  }

  // C/D layout: VGPR r holds (M = half*8 + r, N = l16)
  float* op = out + (size_t)(M0 + wave * 16 + half * 8) * CC + N0 + l16;
#pragma unroll
  for (int r = 0; r < 8; ++r) {
    op[(size_t)r * CC]      = acc0[r];
    op[(size_t)r * CC + 16] = acc1[r];
    op[(size_t)r * CC + 32] = acc2[r];
  }
}

// ---------------- K4: log-softmax denominator per row ----------------
// x[c] = 32 * rinv[row] * raw[row,c] * cinv[c];  denom = max + log(sum exp(x-max))
__global__ void k_lse(const float* __restrict__ raw,
                      const float* __restrict__ rinv,
                      const float* __restrict__ cinv,
                      float* __restrict__ denom) {
  const int r   = blockIdx.x;
  const int tid = threadIdx.x;          // 256 threads
  __shared__ float red[256];
  const float* row = raw + (size_t)r * CC;
  const float scale = 32.0f * rinv[r];
  float xs[6];
  float m = -1e30f;
#pragma unroll
  for (int i = 0; i < 6; ++i) {
    const int c = tid + i * 256;
    if (c < CC) { const float x = scale * row[c] * cinv[c]; xs[i] = x; m = fmaxf(m, x); }
    else        xs[i] = -1e30f;
  }
  red[tid] = m; __syncthreads();
  for (int o = 128; o > 0; o >>= 1) { if (tid < o) red[tid] = fmaxf(red[tid], red[tid + o]); __syncthreads(); }
  m = red[0]; __syncthreads();
  float sum = 0.f;
#pragma unroll
  for (int i = 0; i < 6; ++i) sum += __expf(xs[i] - m);
  red[tid] = sum; __syncthreads();
  for (int o = 128; o > 0; o >>= 1) { if (tid < o) red[tid] += red[tid + o]; __syncthreads(); }
  if (tid == 0) denom[r] = m + logf(red[0]);
}

// ---------------- K5: CTC forward scan, one workgroup per batch element ----------------
__global__ void k_ctc(const float* __restrict__ raw,
                      const float* __restrict__ rinv,
                      const float* __restrict__ cinv,
                      const float* __restrict__ denom,
                      const int* __restrict__ labeling,
                      const int* __restrict__ logit_lgts,
                      const int* __restrict__ labeling_lgts,
                      float* __restrict__ nll) {
  const int n = blockIdx.x;             // 0..31
  const int s = threadIdx.x;            // 64 threads; states 0..60 active
  __shared__ float alpha[2 + LLS];      // alpha[0..1] are permanent NEGINF pads

  // extended label for this state (blank=0 at even s)
  int cl = 0, clm2 = 0;
  if (s < LLS) {
    if (s & 1) cl = labeling[n * SSL + (s >> 1)];
    if (s >= 2 && ((s - 2) & 1)) clm2 = labeling[n * SSL + ((s - 2) >> 1)];
  }
  const bool skip_ok = (s > 1) && (cl != clm2);
  const int  Tlen = logit_lgts[n];

  if (s < 2) alpha[s] = NEG_INF;        // pads, never rewritten
  if (s < LLS) {                        // t = 0 init
    const int row = n;
    const float lp0 = 32.0f * rinv[row] * raw[(size_t)row * CC + cl] * cinv[cl] - denom[row];
    alpha[2 + s] = (s <= 1) ? lp0 : NEG_INF;
  }
  __syncthreads();

  for (int t = 1; t < TT; ++t) {
    float an = NEG_INF;
    if (s < LLS) {
      const float a1 = alpha[2 + s];
      if (t < Tlen) {
        const float a2 = alpha[1 + s];
        const float a3 = skip_ok ? alpha[s] : NEG_INF;
        const int row = t * NNB + n;
        const float lp = 32.0f * rinv[row] * raw[(size_t)row * CC + cl] * cinv[cl] - denom[row];
        const float m = fmaxf(a1, fmaxf(a2, a3));
        an = m + logf(__expf(a1 - m) + __expf(a2 - m) + __expf(a3 - m)) + lp;
      } else {
        an = a1;                        // frozen past this sample's input length
      }
    }
    __syncthreads();
    if (s < LLS) alpha[2 + s] = an;
    __syncthreads();
  }

  if (s == 0) {
    const int Ln = 2 * labeling_lgts[n] + 1;
    const float la = alpha[2 + Ln - 1], lb = alpha[2 + Ln - 2];
    const float m = fmaxf(la, lb);
    nll[n] = -(m + logf(__expf(la - m) + __expf(lb - m)));
  }
}

// ---------------- K6: deterministic fixed-order sum -> scalar ----------------
__global__ void k_sum(const float* __restrict__ nll, float* __restrict__ out) {
  if (threadIdx.x == 0 && blockIdx.x == 0) {
    float sacc = 0.f;
    for (int i = 0; i < NNB; ++i) sacc += nll[i];
    out[0] = sacc;
  }
}

extern "C" void kernel_launch(void* const* d_in, const int* in_sizes, int n_in,
                              void* d_out, int out_size, void* d_ws, size_t ws_size,
                              hipStream_t stream) {
  const float* feats         = (const float*)d_in[0];  // (T*N, D)
  const float* classifier_w  = (const float*)d_in[1];  // (D, C)
  const int*   labeling      = (const int*)d_in[2];    // (N, S)
  const int*   logit_lgts    = (const int*)d_in[3];    // (N,)
  const int*   labeling_lgts = (const int*)d_in[4];    // (N,)
  (void)in_sizes; (void)n_in; (void)out_size; (void)ws_size;

  char* ws = (char*)d_ws;
  size_t off = 0;
  auto alloc = [&](size_t bytes) -> char* {
    char* p = ws + off;
    off = (off + bytes + 255) & ~(size_t)255;
    return p;
  };
  float*  raw   = (float*)alloc((size_t)TT * NNB * CC * sizeof(float));   // 84.9 MB
  __bf16* fB    = (__bf16*)alloc((size_t)TT * NNB * DD * sizeof(__bf16)); // 16.8 MB
  __bf16* wT    = (__bf16*)alloc((size_t)CC * DD * sizeof(__bf16));       //  1.3 MB
  float*  rinv  = (float*)alloc((size_t)TT * NNB * sizeof(float));
  float*  cinv  = (float*)alloc((size_t)CC * sizeof(float));
  float*  denom = (float*)alloc((size_t)TT * NNB * sizeof(float));
  float*  nll   = (float*)alloc((size_t)NNB * sizeof(float));

  k_feat_norm<<<TT * NNB, 32, 0, stream>>>(feats, fB, rinv);
  k_w_norm   <<<CC, 32, 0, stream>>>(classifier_w, wT, cinv);
  k_gemm_wmma<<<dim3(CC / BN, (TT * NNB) / BM), 128, 0, stream>>>(fB, wT, raw);
  k_lse      <<<TT * NNB, 256, 0, stream>>>(raw, rinv, cinv, denom);
  k_ctc      <<<NNB, 64, 0, stream>>>(raw, rinv, cinv, denom,
                                      labeling, logit_lgts, labeling_lgts, nll);
  k_sum      <<<1, 32, 0, stream>>>(nll, (float*)d_out);
}
